// TreeDecoder_3358664426283
// MI455X (gfx1250) — matrix operand order repaired and checked by reading, compiled, not run on hardware
//
#include <hip/hip_runtime.h>
#include <hip/hip_bf16.h>
#include <cmath>

// ---- problem constants (match reference) ----
#define Bsz   256
#define S1    1024
#define Hdim  256
#define Edim  256
#define OPC   50        // OP_NUM + CONST_NUM
#define OPNM1 19        // OP_NUM - 1 (clamp for node_label)
#define NCAND 66        // OPC + NVAR
#define NVARS 16
#define NEGV  (-1e12f)

typedef __bf16 bf16_t;
typedef bf16_t bf16x8  __attribute__((ext_vector_type(8)));
typedef bf16_t bf16x16 __attribute__((ext_vector_type(16)));
typedef float  f32x8   __attribute__((ext_vector_type(8)));

#define LDSP 264   // padded LDS row stride in halves (16B-aligned rows, avoids bank conflicts)

__device__ __forceinline__ float sigmf(float x) { return 1.0f / (1.0f + __expf(-x)); }

// ---------------------------------------------------------------------------
// Kernel 1: current_node = where(has_left, gated(concat_r over [left,cur]),
//                                          gated(concat_l over cur))
// ---------------------------------------------------------------------------
__global__ __launch_bounds__(256) void k_current_node(
    const float* __restrict__ cur, const float* __restrict__ leftc,
    const int* __restrict__ hasleft,
    const float* __restrict__ Wl,  const float* __restrict__ bl,
    const float* __restrict__ Wlg, const float* __restrict__ blg,
    const float* __restrict__ Wr,  const float* __restrict__ br,
    const float* __restrict__ Wrg, const float* __restrict__ brg,
    float* __restrict__ outCN) {
  __shared__ float sc[Hdim], sl[Hdim];
  const int b = blockIdx.x, h = threadIdx.x;
  sc[h] = cur[b * Hdim + h];
  sl[h] = leftc[b * Hdim + h];
  __syncthreads();
  float al = 0.f, alg = 0.f, ar = 0.f, arg = 0.f;
  const float* wl  = Wl  + (size_t)h * Hdim;
  const float* wlg = Wlg + (size_t)h * Hdim;
  const float* wr  = Wr  + (size_t)h * 2 * Hdim;
  const float* wrg = Wrg + (size_t)h * 2 * Hdim;
#pragma unroll 4
  for (int k = 0; k < Hdim; ++k) {
    const float c = sc[k], l = sl[k];
    al  = __fmaf_rn(c, wl[k],  al);
    alg = __fmaf_rn(c, wlg[k], alg);
    ar  = __fmaf_rn(l, wr[k],  ar);            // lc = [left_child, cur]
    arg = __fmaf_rn(l, wrg[k], arg);
    ar  = __fmaf_rn(c, wr[Hdim + k],  ar);
    arg = __fmaf_rn(c, wrg[Hdim + k], arg);
  }
  const float ql = tanhf(al + bl[h]) * sigmf(alg + blg[h]);
  const float qr = tanhf(ar + br[h]) * sigmf(arg + brg[h]);
  outCN[b * Hdim + h] = hasleft[b] ? qr : ql;
}

// ---------------------------------------------------------------------------
// Kernel 2: one-time fp32 -> bf16 conversion of the GEMM weight slices
//   W_enc[n][k] = attn_w.W[n][256+k]   (encoder half of Bahdanau concat)
//   W_sc [e][k] = score_w.W[e][512+k]  (embedding third of score concat)
// ---------------------------------------------------------------------------
__global__ __launch_bounds__(256) void k_convert(
    const float* __restrict__ Wattn, const float* __restrict__ Wsc,
    bf16_t* __restrict__ WencB, bf16_t* __restrict__ WscB) {
  const int idx = blockIdx.x * 256 + threadIdx.x;   // 0..65535
  const int n = idx >> 8, k = idx & 255;
  WencB[idx] = (bf16_t)Wattn[n * 512 + 256 + k];
  WscB[idx]  = (bf16_t)Wsc[n * 768 + 512 + k];
}

// ---------------------------------------------------------------------------
// Kernel 3: curp[b][n] = current_node[b] . attn_w.W[n][0:256] + attn_b[n]
// (the broadcast "hid" half of the attention concat, hoisted out of S)
// ---------------------------------------------------------------------------
__global__ __launch_bounds__(256) void k_curproj(
    const float* __restrict__ cn, const float* __restrict__ Wattn,
    const float* __restrict__ battn, float* __restrict__ curp) {
  __shared__ float sx[Hdim];
  const int b = blockIdx.x, n = threadIdx.x;
  sx[n] = cn[b * Hdim + n];
  __syncthreads();
  const float* w = Wattn + (size_t)n * 2 * Hdim;
  float a = 0.f;
#pragma unroll 4
  for (int k = 0; k < Hdim; ++k) a = __fmaf_rn(sx[k], w[k], a);
  curp[b * Hdim + n] = a + battn[n];
}

// ---------------------------------------------------------------------------
// Kernel 4 (the big one):
//   e[b,s] = score_vec . tanh(enc[b,s,:] @ W_enc^T + curp[b]) + score_b
// One block = 128 s-rows (8 M-tiles) x all 256 N. 8 waves, 2 N-tiles/wave.
// B fragments (the weight matrix) are loaded ONCE per block and held in
// VGPRs (16 x bf16x16 = 128 VGPRs/wave; wave32 allows up to 1024) and reused
// across all 8 M-tiles -> weight L2 traffic drops 8x vs one-tile blocks.
// ---------------------------------------------------------------------------
__global__ __launch_bounds__(256) void k_attn(
    const float* __restrict__ enc, const bf16_t* __restrict__ Wb,
    const float* __restrict__ curp, const float* __restrict__ vs,
    const float* __restrict__ vb, float* __restrict__ e_out) {
  __shared__ __align__(16) bf16_t As[128][LDSP];   // 66 KB bf16 A staging
  __shared__ float red[8][16];
  const int b   = blockIdx.y;
  const int s0  = blockIdx.x * 128;
  const int tid = threadIdx.x;
  const int wv = tid >> 5, lane = tid & 31;
  const int hf = lane >> 4, l16 = lane & 15;
  const int n0 = wv << 5;

  // --- load this wave's B fragments once (kept in registers) ---
  // B layout: lane = col n (l16), 16 contiguous K per lane; low half K=k0..,
  // high half K=k0+16..
  bf16x16 Bf0[8], Bf1[8];
#pragma unroll
  for (int kb = 0; kb < 8; ++kb) {
    const int kk = (kb << 5) + hf * 16;
    Bf0[kb] = *(const bf16x16*)(Wb + (size_t)(n0 + l16) * Hdim + kk);
    Bf1[kb] = *(const bf16x16*)(Wb + (size_t)(n0 + 16 + l16) * Hdim + kk);
  }

  // --- stage 128 x 256 fp32 enc rows as bf16 into LDS ---
  {
    const int c0 = (tid & 15) << 4;
    const int rb = tid >> 4;          // 0..15
#pragma unroll
    for (int rr = 0; rr < 8; ++rr) {
      const int r = rr * 16 + rb;
      const float* src = enc + ((size_t)b * S1 + (s0 + r)) * Hdim + c0;
      bf16_t* dst = &As[r][c0];
#pragma unroll
      for (int i = 0; i < 4; ++i) {
        const float4 f = *(const float4*)(src + i * 4);
        dst[i * 4 + 0] = (bf16_t)f.x; dst[i * 4 + 1] = (bf16_t)f.y;
        dst[i * 4 + 2] = (bf16_t)f.z; dst[i * 4 + 3] = (bf16_t)f.w;
      }
    }
  }
  const int nA = n0 + l16, nB = nA + 16;
  const float cpA = curp[b * Hdim + nA], cpB = curp[b * Hdim + nB];
  const float vA = vs[nA], vB = vs[nB];
  const float bias = vb[0];
  __syncthreads();

  // --- 8 M-tiles, reusing the register-resident B fragments ---
  for (int mt = 0; mt < 8; ++mt) {
    f32x8 acc0 = {}; f32x8 acc1 = {};
#pragma unroll
    for (int kb = 0; kb < 8; ++kb) {
      const int k0 = kb << 5;
      // A frag: lane = row m=l16; halves 0..15 -> K=k0+hf*16+0..15 (contiguous)
      const bf16_t* ap = &As[mt * 16 + l16][k0 + hf * 16];
      const bf16x8 alo = *(const bf16x8*)ap;
      const bf16x8 ahi = *(const bf16x8*)(ap + 8);
      const bf16x16 av = __builtin_shufflevector(alo, ahi,
          0, 1, 2, 3, 4, 5, 6, 7, 8, 9, 10, 11, 12, 13, 14, 15);
      acc0 = __builtin_amdgcn_wmma_f32_16x16x32_bf16(false, av, false, Bf0[kb], (short)0, acc0, false, false);
      acc1 = __builtin_amdgcn_wmma_f32_16x16x32_bf16(false, av, false, Bf1[kb], (short)0, acc1, false, false);
    }
    // epilogue: tanh + dot with attn_score vector, reduce over this wave's 32 n's
#pragma unroll
    for (int r = 0; r < 8; ++r) {   // C layout: VGPR r -> m = r + 8*hf, n = l16
      float p = vA * tanhf(acc0[r] + cpA) + vB * tanhf(acc1[r] + cpB);
      p += __shfl_xor(p, 8, 16);
      p += __shfl_xor(p, 4, 16);
      p += __shfl_xor(p, 2, 16);
      p += __shfl_xor(p, 1, 16);
      if (l16 == 0) red[wv][hf * 8 + r] = p;
    }
    __syncthreads();
    if (tid < 16) {
      float s = bias;
#pragma unroll
      for (int w = 0; w < 8; ++w) s += red[w][tid];
      e_out[(size_t)b * S1 + s0 + mt * 16 + tid] = s;
    }
    __syncthreads();
  }
}

// ---------------------------------------------------------------------------
// Kernel 5: masked softmax over S (in place: e -> attn)
// ---------------------------------------------------------------------------
__global__ __launch_bounds__(256) void k_softmax(
    float* __restrict__ e, const int* __restrict__ lensrc) {
  __shared__ float sred[256];
  const int b = blockIdx.x, tid = threadIdx.x;
  const int len = lensrc[b];
  float v[4]; float mx = -3.0e38f;
#pragma unroll
  for (int i = 0; i < 4; ++i) {
    const int s = tid + (i << 8);
    const float t = (s < len) ? e[(size_t)b * S1 + s] : NEGV;
    v[i] = t; mx = fmaxf(mx, t);
  }
  sred[tid] = mx; __syncthreads();
  for (int off = 128; off; off >>= 1) {
    if (tid < off) sred[tid] = fmaxf(sred[tid], sred[tid + off]);
    __syncthreads();
  }
  mx = sred[0]; __syncthreads();
  float sum = 0.f;
#pragma unroll
  for (int i = 0; i < 4; ++i) { v[i] = __expf(v[i] - mx); sum += v[i]; }
  sred[tid] = sum; __syncthreads();
  for (int off = 128; off; off >>= 1) {
    if (tid < off) sred[tid] += sred[tid + off];
    __syncthreads();
  }
  const float inv = 1.0f / sred[0];
#pragma unroll
  for (int i = 0; i < 4; ++i) e[(size_t)b * S1 + tid + (i << 8)] = v[i] * inv;
}

// ---------------------------------------------------------------------------
// Kernel 6: context[b,h] = sum_s attn[b,s] * enc[b,s,h]
// ---------------------------------------------------------------------------
__global__ __launch_bounds__(256) void k_context(
    const float* __restrict__ enc, const float* __restrict__ attn,
    float* __restrict__ ctx) {
  __shared__ float aw[S1];
  const int b = blockIdx.x, h = threadIdx.x;
#pragma unroll
  for (int i = 0; i < 4; ++i) aw[h + (i << 8)] = attn[(size_t)b * S1 + h + (i << 8)];
  __syncthreads();
  float acc = 0.f;
  const float* ep = enc + (size_t)b * S1 * Hdim + h;
#pragma unroll 4
  for (int s = 0; s < S1; ++s) acc = __fmaf_rn(aw[s], ep[(size_t)s * Hdim], acc);
  ctx[b * Hdim + h] = acc;
}

// ---------------------------------------------------------------------------
// Kernel 7: leaf_proj[b][e] = [cn,ctx] . score_w.W[e][0:512] + score_b[e]
// (broadcast leaf half of the score concat, hoisted out of candidates)
// ---------------------------------------------------------------------------
__global__ __launch_bounds__(256) void k_leafproj(
    const float* __restrict__ cn, const float* __restrict__ ctx,
    const float* __restrict__ Wsc, const float* __restrict__ bsc,
    float* __restrict__ leafp) {
  __shared__ float x1[Hdim], x2[Hdim];
  const int b = blockIdx.x, e = threadIdx.x;
  x1[e] = cn[b * Hdim + e]; x2[e] = ctx[b * Hdim + e];
  __syncthreads();
  const float* w = Wsc + (size_t)e * 768;
  float a = 0.f;
#pragma unroll 4
  for (int k = 0; k < Hdim; ++k) {
    a = __fmaf_rn(x1[k], w[k], a);
    a = __fmaf_rn(x2[k], w[Hdim + k], a);
  }
  leafp[b * Edim + e] = a + bsc[e];
}

// ---------------------------------------------------------------------------
// Kernel 8: candidate scores via bf16 WMMA.
// Rows r = b*66 + c (16896 rows = 1056 tiles of 16). A-row = op/const emb or
// gathered encoder state; N = 256 emb dims; epilogue = tanh + score_out dot + mask.
// ---------------------------------------------------------------------------
__global__ __launch_bounds__(256) void k_score(
    const float* __restrict__ enc, const float* __restrict__ embtab,
    const int* __restrict__ varpos, const int* __restrict__ lenvar,
    const bf16_t* __restrict__ Wb, const float* __restrict__ leafp,
    const float* __restrict__ vo, const float* __restrict__ vob,
    float* __restrict__ nscore) {
  __shared__ __align__(16) bf16_t As[16][LDSP];
  __shared__ float red[8][16];
  const int r0  = blockIdx.x * 16;
  const int tid = threadIdx.x;
  {
    const int r   = tid >> 4;
    const int row = r0 + r;
    const int bb  = row / NCAND;
    const int c   = row - bb * NCAND;
    const float* src = (c < OPC)
        ? (embtab + (size_t)c * Edim)
        : (enc + ((size_t)bb * S1 + varpos[bb * NVARS + (c - OPC)]) * Hdim);
    const int c0 = (tid & 15) << 4;
    bf16_t* dst = &As[r][c0];
#pragma unroll
    for (int i = 0; i < 4; ++i) {
      const float4 f = *(const float4*)(src + c0 + i * 4);
      dst[i * 4 + 0] = (bf16_t)f.x; dst[i * 4 + 1] = (bf16_t)f.y;
      dst[i * 4 + 2] = (bf16_t)f.z; dst[i * 4 + 3] = (bf16_t)f.w;
    }
  }
  __syncthreads();
  const int wv = tid >> 5, lane = tid & 31;
  const int hf = lane >> 4, l16 = lane & 15;
  const int n0 = wv << 5;
  f32x8 acc0 = {}; f32x8 acc1 = {};
#pragma unroll
  for (int kb = 0; kb < 8; ++kb) {
    const int k0 = kb << 5;
    const bf16_t* ap = &As[l16][k0 + hf * 16];
    const bf16x8 alo = *(const bf16x8*)ap;
    const bf16x8 ahi = *(const bf16x8*)(ap + 8);
    const bf16x16 av = __builtin_shufflevector(alo, ahi,
        0, 1, 2, 3, 4, 5, 6, 7, 8, 9, 10, 11, 12, 13, 14, 15);
    const int kk = k0 + hf * 16;
    const bf16x16 bv0 = *(const bf16x16*)(Wb + (size_t)(n0 + l16) * Edim + kk);
    const bf16x16 bv1 = *(const bf16x16*)(Wb + (size_t)(n0 + 16 + l16) * Edim + kk);
    acc0 = __builtin_amdgcn_wmma_f32_16x16x32_bf16(false, av, false, bv0, (short)0, acc0, false, false);
    acc1 = __builtin_amdgcn_wmma_f32_16x16x32_bf16(false, av, false, bv1, (short)0, acc1, false, false);
  }
  const int nA = n0 + l16, nB = nA + 16;
  const float vA = vo[nA], vB = vo[nB];
#pragma unroll
  for (int r = 0; r < 8; ++r) {
    const int row = r0 + hf * 8 + r;
    const int bb  = row / NCAND;
    const float cpA = leafp[bb * Edim + nA];
    const float cpB = leafp[bb * Edim + nB];
    float p = vA * tanhf(acc0[r] + cpA) + vB * tanhf(acc1[r] + cpB);
    p += __shfl_xor(p, 8, 16);
    p += __shfl_xor(p, 4, 16);
    p += __shfl_xor(p, 2, 16);
    p += __shfl_xor(p, 1, 16);
    if (l16 == 0) red[wv][hf * 8 + r] = p;
  }
  __syncthreads();
  if (tid < 16) {
    const int row = r0 + tid;
    const int bb  = row / NCAND;
    const int c   = row - bb * NCAND;
    float s = vob[0];
#pragma unroll
    for (int w = 0; w < 8; ++w) s += red[w][tid];
    nscore[row] = (c < lenvar[bb] + OPC) ? s : NEGV;
  }
}

// ---------------------------------------------------------------------------
// Kernel 9: GenerateNode — gated l_child / r_child from gcat=[cn,ctx,tok]
// ---------------------------------------------------------------------------
__global__ __launch_bounds__(256) void k_gen(
    const float* __restrict__ cn, const float* __restrict__ ctx,
    const float* __restrict__ embtab, const int* __restrict__ label,
    const float* __restrict__ Wl,  const float* __restrict__ bl,
    const float* __restrict__ Wlg, const float* __restrict__ blg,
    const float* __restrict__ Wr,  const float* __restrict__ br,
    const float* __restrict__ Wrg, const float* __restrict__ brg,
    float* __restrict__ lout, float* __restrict__ rout) {
  __shared__ float x[768];
  const int b = blockIdx.x, h = threadIdx.x;
  const int lab = (label[b] < OPNM1) ? label[b] : OPNM1;
  x[h]            = cn[b * Hdim + h];
  x[Hdim + h]     = ctx[b * Hdim + h];
  x[2 * Hdim + h] = embtab[(size_t)lab * Edim + h];
  __syncthreads();
  float al = 0.f, alg = 0.f, ar = 0.f, arg = 0.f;
  const float* wl  = Wl  + (size_t)h * 768;
  const float* wlg = Wlg + (size_t)h * 768;
  const float* wr  = Wr  + (size_t)h * 768;
  const float* wrg = Wrg + (size_t)h * 768;
#pragma unroll 4
  for (int k = 0; k < 768; ++k) {
    const float xv = x[k];
    al  = __fmaf_rn(xv, wl[k],  al);
    alg = __fmaf_rn(xv, wlg[k], alg);
    ar  = __fmaf_rn(xv, wr[k],  ar);
    arg = __fmaf_rn(xv, wrg[k], arg);
  }
  lout[b * Hdim + h] = tanhf(al + bl[h]) * sigmf(alg + blg[h]);
  rout[b * Hdim + h] = tanhf(ar + br[h]) * sigmf(arg + brg[h]);
}

// ---------------------------------------------------------------------------
// Kernel 10: Merge — sub_tree from mcat=[tok, l_child, r_child]
// ---------------------------------------------------------------------------
__global__ __launch_bounds__(256) void k_merge(
    const float* __restrict__ embtab, const int* __restrict__ label,
    const float* __restrict__ lch, const float* __restrict__ rch,
    const float* __restrict__ Wm,  const float* __restrict__ bm,
    const float* __restrict__ Wmg, const float* __restrict__ bmg,
    float* __restrict__ stout) {
  __shared__ float x[768];
  const int b = blockIdx.x, h = threadIdx.x;
  const int lab = (label[b] < OPNM1) ? label[b] : OPNM1;
  x[h]            = embtab[(size_t)lab * Edim + h];
  x[Hdim + h]     = lch[b * Hdim + h];
  x[2 * Hdim + h] = rch[b * Hdim + h];
  __syncthreads();
  float am = 0.f, amg = 0.f;
  const float* wm  = Wm  + (size_t)h * 768;
  const float* wmg = Wmg + (size_t)h * 768;
#pragma unroll 4
  for (int k = 0; k < 768; ++k) {
    const float xv = x[k];
    am  = __fmaf_rn(xv, wm[k],  am);
    amg = __fmaf_rn(xv, wmg[k], amg);
  }
  stout[b * Hdim + h] = tanhf(am + bm[h]) * sigmf(amg + bmg[h]);
}

// ---------------------------------------------------------------------------
extern "C" void kernel_launch(void* const* d_in, const int* in_sizes, int n_in,
                              void* d_out, int out_size, void* d_ws, size_t ws_size,
                              hipStream_t stream) {
  (void)in_sizes; (void)n_in; (void)out_size;
  // params flattened in setup_inputs() dict order (W,b per tuple):
  const float* emb    = (const float*)d_in[0];
  const float* Wcl    = (const float*)d_in[1];
  const float* bcl    = (const float*)d_in[2];
  const float* Wcr    = (const float*)d_in[3];
  const float* bcr    = (const float*)d_in[4];
  const float* Wclg   = (const float*)d_in[5];
  const float* bclg   = (const float*)d_in[6];
  const float* Wcrg   = (const float*)d_in[7];
  const float* bcrg   = (const float*)d_in[8];
  const float* Wattn  = (const float*)d_in[9];
  const float* battn  = (const float*)d_in[10];
  const float* Wasc   = (const float*)d_in[11];
  const float* basc   = (const float*)d_in[12];
  const float* Wsc    = (const float*)d_in[13];
  const float* bsc    = (const float*)d_in[14];
  const float* Wso    = (const float*)d_in[15];
  const float* bso    = (const float*)d_in[16];
  const float* Wgl    = (const float*)d_in[17];
  const float* bgl    = (const float*)d_in[18];
  const float* Wgr    = (const float*)d_in[19];
  const float* bgr    = (const float*)d_in[20];
  const float* Wglg   = (const float*)d_in[21];
  const float* bglg   = (const float*)d_in[22];
  const float* Wgrg   = (const float*)d_in[23];
  const float* bgrg   = (const float*)d_in[24];
  const float* Wm     = (const float*)d_in[25];
  const float* bm     = (const float*)d_in[26];
  const float* Wmg    = (const float*)d_in[27];
  const float* bmg    = (const float*)d_in[28];
  const float* enc    = (const float*)d_in[29];
  const float* curemb = (const float*)d_in[30];
  const float* leftch = (const float*)d_in[31];
  const int* hasleft  = (const int*)d_in[32];
  const int* lensrc   = (const int*)d_in[33];
  const int* varpos   = (const int*)d_in[34];
  const int* lenvar   = (const int*)d_in[35];
  const int* nodelab  = (const int*)d_in[36];

  // output layout (concatenated tuple, fp32)
  float* out      = (float*)d_out;
  float* nscore   = out;                       // [B,66]
  float* cn_out   = out + 16896;               // [B,H]
  float* ctx_out  = out + 82432;               // [B,H]
  float* l_out    = out + 147968;              // [B,H]
  float* r_out    = out + 213504;              // [B,H]
  float* st_out   = out + 279040;              // [B,H]

  // workspace layout
  if (ws_size < (size_t)1835008) return;
  char* w = (char*)d_ws;
  bf16_t* WencB = (bf16_t*)(w);                // 128 KB  bf16 [256][256]
  bf16_t* WscB  = (bf16_t*)(w + 131072);       // 128 KB  bf16 [256][256]
  float*  curp  = (float*)(w + 262144);        // 256 KB  [B,H]
  float*  ebuf  = (float*)(w + 524288);        // 1 MB    [B,S] e -> attn (in place)
  float*  leafp = (float*)(w + 1572864);       // 256 KB  [B,E]

  k_current_node<<<Bsz, 256, 0, stream>>>(curemb, leftch, hasleft,
      Wcl, bcl, Wclg, bclg, Wcr, bcr, Wcrg, bcrg, cn_out);
  k_convert<<<256, 256, 0, stream>>>(Wattn, Wsc, WencB, WscB);
  k_curproj<<<Bsz, 256, 0, stream>>>(cn_out, Wattn, battn, curp);
  k_attn<<<dim3(S1 / 128, Bsz), 256, 0, stream>>>(enc, WencB, curp, Wasc, basc, ebuf);
  k_softmax<<<Bsz, 256, 0, stream>>>(ebuf, lensrc);
  k_context<<<Bsz, 256, 0, stream>>>(enc, ebuf, ctx_out);
  k_leafproj<<<Bsz, 256, 0, stream>>>(cn_out, ctx_out, Wsc, bsc, leafp);
  k_score<<<(Bsz * NCAND) / 16, 256, 0, stream>>>(enc, emb, varpos, lenvar,
      WscB, leafp, Wso, bso, nscore);
  k_gen<<<Bsz, 256, 0, stream>>>(cn_out, ctx_out, emb, nodelab,
      Wgl, bgl, Wglg, bglg, Wgr, bgr, Wgrg, bgrg, l_out, r_out);
  k_merge<<<Bsz, 256, 0, stream>>>(emb, nodelab, l_out, r_out,
      Wm, bm, Wmg, bmg, st_out);
}
